// DPMMatchingHead_28518582845737
// MI455X (gfx1250) — matrix operand order repaired and checked by reading, not compiled, not run on hardware
//
#include <hip/hip_runtime.h>
#include <hip/hip_bf16.h>

// Problem constants (from reference setup_inputs)
constexpr int B = 64, K = 16, N = 4096, C = 384, M = 4096;
#define EPSF 1e-8f

typedef __attribute__((ext_vector_type(16))) __bf16 v16bf;
typedef __attribute__((ext_vector_type(8)))  float  v8f;

union BF16x16 { v16bf v; uint4 q[2]; unsigned short s[16]; };

static __device__ __forceinline__ unsigned short f2bf(float f) {
  unsigned u = __builtin_bit_cast(unsigned, f);
  u += 0x7FFFu + ((u >> 16) & 1u);           // round-to-nearest-even
  return (unsigned short)(u >> 16);
}

// ---------------------------------------------------------------------------
// Stage fp32 rows -> bf16 rows + row L2 norms. One wave per row of C=384.
// Used for feats (once), feats_org (once), and prototypes (per iteration).
// ---------------------------------------------------------------------------
__global__ void prep_bf16(const float* __restrict__ src, unsigned short* __restrict__ dst,
                          float* __restrict__ norms, int nrows) {
  int wid = threadIdx.x >> 5, lane = threadIdx.x & 31;
  int row = blockIdx.x * 8 + wid;
  if (row >= nrows) return;
  const float* p = src + (size_t)row * C;
  unsigned short* q = dst + (size_t)row * C;
  float s = 0.f;
#pragma unroll
  for (int j = 0; j < 3; ++j) {
    int c = j * 128 + lane * 4;
    float4 v = *(const float4*)(p + c);
    s += v.x * v.x + v.y * v.y + v.z * v.z + v.w * v.w;
    uint2 o;
    o.x = (unsigned)f2bf(v.x) | ((unsigned)f2bf(v.y) << 16);
    o.y = (unsigned)f2bf(v.z) | ((unsigned)f2bf(v.w) << 16);
    *(uint2*)(q + c) = o;
  }
  s += __shfl_xor(s, 1);  s += __shfl_xor(s, 2);
  s += __shfl_xor(s, 4);  s += __shfl_xor(s, 8);  s += __shfl_xor(s, 16);
  if (lane == 0) norms[row] = sqrtf(s);
}

__global__ void init_state(float* tau, float* dens, float* msum) {
  int i = blockIdx.x * 256 + threadIdx.x;
  if (i < B * K) { tau[i] = 0.1f; dens[i] = 0.f; msum[i] = 0.f; }
}

// ---------------------------------------------------------------------------
// sim kernel: per-batch [K=16 x C=384] x [C x 16-wide n tile] cosine GEMM.
// A and B both pre-staged bf16 -> pure b128 loads + WMMA in the loop.
// MODE 0: store sim only.  MODE 1: store sim AND accumulate masked density.
// WMMA layouts (wave32):
//   A 16x32 bf16 : lane l, row m=l&15; elems j -> K=c0+half*16+j
//                  (contiguous)                             (half=l>>4)
//   B 32x16 bf16 : lane col n=l&15; elems j -> K=c0+half*16+j (contiguous)
//   C/D 16x16 f32: VGPR r -> m = half*8 + r, n = l&15
// ---------------------------------------------------------------------------
template <int MODE>
__global__ void sim_kernel(const unsigned short* __restrict__ pbf,
                           const unsigned short* __restrict__ fbf,
                           const float* __restrict__ pnorm, const float* __restrict__ fnorm,
                           float* __restrict__ simbuf, const int* __restrict__ idx,
                           float* __restrict__ dens) {
  int wid = threadIdx.x >> 5, lane = threadIdx.x & 31;
  int tile = blockIdx.x * 8 + wid;          // B * (N/16) tiles
  int b = tile >> 8, t = tile & 255;
  int n0 = t * 16;
  int half = lane >> 4, lr = lane & 15;
  const unsigned short* arow = pbf + ((size_t)b * K + lr) * C;
  const unsigned short* brow = fbf + ((size_t)b * N + n0 + lr) * C;
  v8f acc = {};
  for (int c0 = 0; c0 < C; c0 += 32) {
    BF16x16 A, Bm;
    A.q[0]  = *(const uint4*)(arow + c0 + half * 16);
    A.q[1]  = *(const uint4*)(arow + c0 + half * 16 + 8);
    const uint4* pb = (const uint4*)(brow + c0 + half * 16);
    Bm.q[0] = pb[0]; Bm.q[1] = pb[1];
    acc = __builtin_amdgcn_wmma_f32_16x16x32_bf16(false, A.v, false, Bm.v,
                                                  (short)0, acc, false, false);
  }
  float fn = fnorm[(size_t)b * N + n0 + lr];
  int myidx = (MODE == 1) ? idx[(size_t)b * N + n0 + lr] : -1;
#pragma unroll
  for (int r = 0; r < 8; ++r) {
    int m = half * 8 + r;
    float val = acc[r] / fmaxf(pnorm[b * K + m] * fn, EPSF);
    simbuf[((size_t)(b * K + m)) * N + n0 + lr] = val;
    if (MODE == 1) {
      float contrib = (myidx == m) ? val : 0.f;
      contrib += __shfl_xor(contrib, 1);
      contrib += __shfl_xor(contrib, 2);
      contrib += __shfl_xor(contrib, 4);
      contrib += __shfl_xor(contrib, 8);
      if (lr == 0) atomicAdd(&dens[b * K + m], contrib);
    }
  }
}

// ---------------------------------------------------------------------------
// Softmax row stats over N per (b,k): max and sum of exp((x-max)*invtt).
// ---------------------------------------------------------------------------
__global__ void softmax_stats(const float* __restrict__ sim, const float* __restrict__ tau,
                              float* __restrict__ rmax, float* __restrict__ rsum) {
  __shared__ float red[256];
  int bk = blockIdx.x, tid = threadIdx.x;
  const float* row = sim + (size_t)bk * N;
  float m = -1e30f;
  for (int n = tid; n < N; n += 256) m = fmaxf(m, row[n]);
  red[tid] = m; __syncthreads();
  for (int s = 128; s > 0; s >>= 1) { if (tid < s) red[tid] = fmaxf(red[tid], red[tid + s]); __syncthreads(); }
  m = red[0]; __syncthreads();
  float invtt = 1.f / (0.1f * tau[bk]);
  float sum = 0.f;
  for (int n = tid; n < N; n += 256) sum += __expf((row[n] - m) * invtt);
  red[tid] = sum; __syncthreads();
  for (int s = 128; s > 0; s >>= 1) { if (tid < s) red[tid] += red[tid + s]; __syncthreads(); }
  if (tid == 0) { rmax[bk] = m; rsum[bk] = red[0]; }
}

// Per column n: softmax weights, argmax over K (first max), masked weight in
// bf16 (A-matrix of the update GEMM), assignment index, and per-k counts.
__global__ void mask_kernel(const float* __restrict__ sim, const float* __restrict__ rmax,
                            const float* __restrict__ rsum, const float* __restrict__ tau,
                            unsigned short* __restrict__ wmbf, int* __restrict__ idx,
                            float* __restrict__ msum) {
  int gid = blockIdx.x * 256 + threadIdx.x;   // B*N
  int b = gid >> 12, n = gid & (N - 1);
  float w16[K]; int best = 0; float bw = -1.f;
#pragma unroll
  for (int k = 0; k < K; ++k) {
    int bk = b * K + k;
    float x = sim[(size_t)bk * N + n];
    float wv = __expf((x - rmax[bk]) * (1.f / (0.1f * tau[bk]))) / rsum[bk];
    w16[k] = wv;
    if (wv > bw) { bw = wv; best = k; }   // strict > keeps first occurrence
  }
#pragma unroll
  for (int k = 0; k < K; ++k)
    wmbf[(size_t)(b * K + k) * N + n] = (k == best) ? f2bf(w16[k]) : (unsigned short)0;
  idx[gid] = best;
  atomicAdd(&msum[b * K + best], 1.f);
}

// ---------------------------------------------------------------------------
// Prototype update: proto[b,k,c] = sum_n wm[b,k,n] * feats[b,n,c].
// One block per batch, 24 waves (one 16-wide c-tile each). The [32 x 384]
// bf16 feats tiles are staged global->LDS with GLOBAL_LOAD_ASYNC_TO_LDS_B128
// into a DOUBLE BUFFER: the next tile's async copy is in flight while the
// current tile is consumed. Async loads complete in order, so after issuing
// the 2 next-tile loads, `s_wait_asynccnt 2` guarantees the 2 older
// (current-tile) loads have landed.
// ---------------------------------------------------------------------------
__global__ void __launch_bounds__(768)
proto_update(const unsigned short* __restrict__ wmbf, const unsigned short* __restrict__ fbf,
             float* __restrict__ pout) {
  __shared__ unsigned short tile[2][32 * C];   // 2 x 24 KB
  int b = blockIdx.x, tid = threadIdx.x;
  int wid = tid >> 5, lane = tid & 31;
  int half = lane >> 4, lr = lane & 15;
  int c0 = wid * 16;
  const unsigned short* wrow = wmbf + ((size_t)b * K + lr) * N;
  int ldr = tid / 24, ldc = (tid % 24) * 16;      // 32 rows x 24 chunks = 768 thr
  const unsigned short* gsrc = fbf + ((size_t)b * N + ldr) * C + ldc;
  // LDS byte addresses of this thread's 32-byte staging slot in each buffer
  // (low 32 bits of a generic LDS pointer == offset in the wave's LDS alloc).
  unsigned lds0 = (unsigned)(uintptr_t)&tile[0][ldr * C + ldc];
  unsigned lds1 = (unsigned)(uintptr_t)&tile[1][ldr * C + ldc];

  v8f acc = {};
  // Prologue: kick off tile 0 into buffer 0.
  {
    const unsigned short* g = gsrc;
    asm volatile("global_load_async_to_lds_b128 %0, %1, off"
                 :: "v"(lds0), "v"(g) : "memory");
    asm volatile("global_load_async_to_lds_b128 %0, %1, off offset:16"
                 :: "v"(lds0), "v"(g) : "memory");
  }
  for (int nb = 0; nb < N; nb += 32) {
    int cur = (nb >> 5) & 1;
    // Issue next tile into the other buffer (safe: previous iteration's
    // trailing barrier guarantees nobody is still reading it).
    if (nb + 32 < N) {
      const unsigned short* g = gsrc + (size_t)(nb + 32) * C;
      unsigned dst = cur ? lds0 : lds1;
      asm volatile("global_load_async_to_lds_b128 %0, %1, off"
                   :: "v"(dst), "v"(g) : "memory");
      asm volatile("global_load_async_to_lds_b128 %0, %1, off offset:16"
                   :: "v"(dst), "v"(g) : "memory");
      asm volatile("s_wait_asynccnt 2" ::: "memory");   // current tile landed
    } else {
      asm volatile("s_wait_asynccnt 0" ::: "memory");
    }
    __syncthreads();                        // current tile visible block-wide
    const unsigned short* tb = &tile[cur][0];
    BF16x16 A, Bm;
    A.q[0] = *(const uint4*)(wrow + nb + half * 16);
    A.q[1] = *(const uint4*)(wrow + nb + half * 16 + 8);
#pragma unroll
    for (int j = 0; j < 16; ++j)
      Bm.s[j] = tb[(half * 16 + j) * C + c0 + lr];   // ds_load_u16 gathers
    acc = __builtin_amdgcn_wmma_f32_16x16x32_bf16(false, A.v, false, Bm.v,
                                                  (short)0, acc, false, false);
    __syncthreads();                        // all waves done with `cur` buffer
  }
#pragma unroll
  for (int r = 0; r < 8; ++r)
    pout[((size_t)b * K + half * 8 + r) * C + c0 + lr] = acc[r];
}

__global__ void tau_update(float* __restrict__ tau, float* __restrict__ dens,
                           float* __restrict__ msum) {
  int i = blockIdx.x * 256 + threadIdx.x;
  if (i < B * K) {
    float ms = msum[i], d = dens[i];
    float dv = (ms >= 1.f) ? (d / fmaxf(ms, 1.f)) : 0.f;
    tau[i] = fmaxf(1.f - dv, 1e-10f);
    dens[i] = 0.f; msum[i] = 0.f;            // ready for next iteration
  }
}

// Final cosine GEMM: [1024 x 384] x [384 x 4096] vs feats_org (both bf16).
__global__ void final_sim(const unsigned short* __restrict__ pbf,
                          const unsigned short* __restrict__ obf,
                          const float* __restrict__ pnorm, const float* __restrict__ onorm,
                          float* __restrict__ out) {
  int wid = threadIdx.x >> 5, lane = threadIdx.x & 31;
  int tile = blockIdx.x * 8 + wid;           // 64 row-tiles * 256 m-tiles
  int rt = tile >> 8, mt = tile & 255;
  int m0 = mt * 16;
  int half = lane >> 4, lr = lane & 15;
  const unsigned short* arow = pbf + ((size_t)rt * 16 + lr) * C;
  const unsigned short* brow = obf + (size_t)(m0 + lr) * C;
  v8f acc = {};
  for (int c0 = 0; c0 < C; c0 += 32) {
    BF16x16 A, Bm;
    A.q[0]  = *(const uint4*)(arow + c0 + half * 16);
    A.q[1]  = *(const uint4*)(arow + c0 + half * 16 + 8);
    const uint4* pb = (const uint4*)(brow + c0 + half * 16);
    Bm.q[0] = pb[0]; Bm.q[1] = pb[1];
    acc = __builtin_amdgcn_wmma_f32_16x16x32_bf16(false, A.v, false, Bm.v,
                                                  (short)0, acc, false, false);
  }
  float on = onorm[m0 + lr];
#pragma unroll
  for (int r = 0; r < 8; ++r) {
    int row = rt * 16 + half * 8 + r;
    out[(size_t)row * M + m0 + lr] = acc[r] / fmaxf(pnorm[row] * on, EPSF);
  }
}

__global__ void copy_f32(const float* __restrict__ s, float* __restrict__ d, int n) {
  int i = blockIdx.x * 256 + threadIdx.x;
  if (i < n) d[i] = s[i];
}

// ---------------------------------------------------------------------------
extern "C" void kernel_launch(void* const* d_in, const int* in_sizes, int n_in,
                              void* d_out, int out_size, void* d_ws, size_t ws_size,
                              hipStream_t stream) {
  const float* proto_in = (const float*)d_in[0];   // [B,K,C]
  const float* feats    = (const float*)d_in[1];   // [B,N,C]
  const float* forg     = (const float*)d_in[2];   // [M,C]

  char* w = (char*)d_ws;
  size_t off = 0;
  auto alloc = [&](size_t bytes) { void* p = w + off; off = (off + bytes + 255) & ~(size_t)255; return p; };

  unsigned short* featsbf = (unsigned short*)alloc((size_t)B * N * C * 2);  // 201 MB
  unsigned short* orgbf   = (unsigned short*)alloc((size_t)M * C * 2);
  unsigned short* protobf = (unsigned short*)alloc((size_t)B * K * C * 2);
  unsigned short* wmbf    = (unsigned short*)alloc((size_t)B * K * N * 2);  // 8 MB
  float* fnorm  = (float*)alloc((size_t)B * N * 4);
  float* onorm  = (float*)alloc((size_t)M * 4);
  float* simbuf = (float*)alloc((size_t)B * K * N * 4);                     // 16 MB
  float* pA     = (float*)alloc((size_t)B * K * C * 4);
  float* pB     = (float*)alloc((size_t)B * K * C * 4);
  float* pnorm  = (float*)alloc(B * K * 4);
  float* rmax   = (float*)alloc(B * K * 4);
  float* rsum   = (float*)alloc(B * K * 4);
  float* tau    = (float*)alloc(B * K * 4);
  float* dens   = (float*)alloc(B * K * 4);
  float* msum   = (float*)alloc(B * K * 4);
  int*   idx    = (int*)alloc((size_t)B * N * 4);

  float* out_proto = (float*)d_out;                 // [B*K, C]
  float* out_sim   = out_proto + (size_t)B * K * C; // [B*K, M]

  // Stage bf16 copies + norms (amortizes fp32 HBM traffic over all passes)
  prep_bf16<<<(B * N + 7) / 8, 256, 0, stream>>>(feats, featsbf, fnorm, B * N);
  prep_bf16<<<(M + 7) / 8, 256, 0, stream>>>(forg, orgbf, onorm, M);
  init_state<<<4, 256, 0, stream>>>(tau, dens, msum);

  // Stage initial prototypes (bf16 + norms), then initial similarity map.
  prep_bf16<<<(B * K + 7) / 8, 256, 0, stream>>>(proto_in, protobf, pnorm, B * K);
  sim_kernel<0><<<(B * (N / 16)) / 8, 256, 0, stream>>>(protobf, featsbf, pnorm, fnorm,
                                                        simbuf, nullptr, nullptr);

  const float* cur = proto_in;
  for (int it = 0; it < 5; ++it) {
    softmax_stats<<<B * K, 256, 0, stream>>>(simbuf, tau, rmax, rsum);
    mask_kernel<<<(B * N) / 256, 256, 0, stream>>>(simbuf, rmax, rsum, tau, wmbf, idx, msum);
    float* nxt = (it & 1) ? pB : pA;
    proto_update<<<B, 768, 0, stream>>>(wmbf, featsbf, nxt);
    prep_bf16<<<(B * K + 7) / 8, 256, 0, stream>>>(nxt, protobf, pnorm, B * K);
    if (it < 4) {
      // sim2 of iter `it` == sim_map of iter `it+1`: compute once, fused with density
      sim_kernel<1><<<(B * (N / 16)) / 8, 256, 0, stream>>>(protobf, featsbf, pnorm, fnorm,
                                                            simbuf, idx, dens);
      tau_update<<<4, 256, 0, stream>>>(tau, dens, msum);
    }
    cur = nxt;
  }

  copy_f32<<<(B * K * C + 255) / 256, 256, 0, stream>>>(cur, out_proto, B * K * C);
  final_sim<<<((B * K / 16) * (M / 16)) / 8, 256, 0, stream>>>(protobf, orgbf, pnorm, onorm, out_sim);
}
